// DQN_57380763075162
// MI455X (gfx1250) — compile-verified, matrix operand order checked by reading, not run on hardware
//
#include <hip/hip_runtime.h>
#include <hip/hip_bf16.h>
#include <math.h>

// CDNA5 (gfx1250) wave32 WMMA types
typedef __attribute__((ext_vector_type(16))) _Float16 v16h;
typedef __attribute__((ext_vector_type(8)))  _Float16 v8h;
typedef __attribute__((ext_vector_type(8)))  float    v8f;

#define MCAND     50     // CEM candidates per state
#define MPAD      64     // padded rows: 4 waves x 16
#define NTOP      32     // elites
#define CEM_ITERS 100    // reference ITERS (99 evaluated steps)
#define HID       100
#define NPAD      112    // 7 * 16  (N tiles)
#define KPAD      128    // 4 * 32  (K steps)
#define NT        7
#define KT        4

__device__ __forceinline__ unsigned pcg_hash(unsigned x) {
  x = x * 747796405u + 2891336453u;
  unsigned w = ((x >> ((x >> 28u) + 4u)) ^ x) * 277803737u;
  return (w >> 22u) ^ w;
}
__device__ __forceinline__ float u01(unsigned h) {
  return (float)(h >> 8) * (1.0f / 16777216.0f);
}
// A-fragment K permutation: within each 32-wide K block store order
// [0..7, 16..23, 8..15, 24..31]  (= swap bits 3 and 4; an involution that
// maps 8-aligned runs to 8-aligned runs, so vector stores stay contiguous).
__device__ __forceinline__ int kperm(int k) {
  return (k & ~0x18) | ((k & 8) << 1) | ((k & 16) >> 1);
}

__global__ __launch_bounds__(128)
void cem_dqn_kernel(const float* __restrict__ states,
                    const float* __restrict__ W1g, const float* __restrict__ b1g,
                    const float* __restrict__ W2g, const float* __restrict__ b2g,
                    const float* __restrict__ W3g, const float* __restrict__ b3g,
                    float* __restrict__ out)
{
  // Weights pre-swizzled into per-lane WMMA fragment order (f16), resident in
  // LDS for all 99 iterations. sHp holds activations in K-permuted, row-XOR-
  // swizzled layout: fragment loads/stores are single contiguous vector ops
  // and 8 consecutive rows hit disjoint LDS bank groups.
  __shared__ __align__(32) _Float16 sW1Tp[NT][32][16];       // layer1 A (W1^T+b1)
  __shared__ __align__(32) _Float16 sW2p[NT][KT][32][16];    // layer2 B
  __shared__ __align__(32) _Float16 sHp[MPAD][KPAD];         // activations
  __shared__ float sB2[NPAD], sW3[NPAD];
  __shared__ float sAng[MPAD], sQ[MPAD];
  __shared__ float sMu, sStd, sSum, sSumSq;

  const int tid    = threadIdx.x;
  const int lane   = tid & 31;
  const int wave   = tid >> 5;
  const int base   = wave * 16;          // this wave's 16 candidate rows
  const unsigned s = blockIdx.x;

  __builtin_prefetch(W2g, 0, 3);         // global_prefetch_b8: warm L2/WGP$

  // ---- one-time staging: f32 -> f16, packed into fragment order ----
  // Layer 1 runs transposed: D(M=hidden, N=cand) = A(W1^T, M=hidden,K=4) x
  // B(X^T, K=4,N=cand). Bias folds in as K=3 with B[3][*]=1.
  // A f16 16x32 layout: lane l holds row M=(l&15), K {klo..klo+7, 16+klo..},
  // klo=(l>>4)*8.
  for (int idx = tid; idx < NT * 32 * 16; idx += 128) {
    int e = idx & 15, l = (idx >> 4) & 31, t = idx >> 9;
    int n   = t * 16 + (l & 15);         // hidden unit (M)
    int klo = (l >> 4) * 8;
    int k   = (e < 8) ? (klo + e) : (16 + klo + (e - 8));
    float v = 0.0f;
    if (n < HID) {
      if      (k == 0) v = W1g[n];
      else if (k == 1) v = W1g[HID + n];
      else if (k == 2) v = W1g[2 * HID + n];
      else if (k == 3) v = b1g[n];       // bias lane
    }
    sW1Tp[t][l][e] = (_Float16)v;
  }
  // Layer 2 B fragments: lane l holds col = 16*t + (l&15), same K pattern.
  for (int idx = tid; idx < NT * KT * 32 * 16; idx += 128) {
    int e = idx & 15, l = (idx >> 4) & 31, ks = (idx >> 9) & 3, t = idx >> 11;
    int col = t * 16 + (l & 15);
    int klo = (l >> 4) * 8;
    int k   = ks * 32 + ((e < 8) ? (klo + e) : (16 + klo + (e - 8)));
    float v = (k < HID && col < HID) ? W2g[k * HID + col] : 0.0f;
    sW2p[t][ks][l][e] = (_Float16)v;
  }
  for (int n = tid; n < NPAD; n += 128) {
    sB2[n] = (n < HID) ? b2g[n] : 0.0f;
    sW3[n] = (n < HID) ? W3g[n] : 0.0f;
  }
  for (int idx = tid; idx < MPAD * KPAD; idx += 128)
    (&sHp[0][0])[idx] = (_Float16)0.0f;  // zero K-padding once

  const float sx = states[s * 2 + 0];
  const float sy = states[s * 2 + 1];
  const float B3 = b3g[0];
  __syncthreads();

  const int nloc   = lane & 15;          // tile-local N index
  const int half   = lane >> 4;          // lane half selects K sub-block
  const int hrow   = base + nloc;        // this lane's candidate row in sHp
  const int rxor   = (hrow & 7);         // LDS bank swizzle for sHp groups

  for (int it = 0; it < CEM_ITERS - 1; ++it) {
    // ---- sample candidate "angles" (iter 0: U[0,1); else Normal(mu,std)) ----
    if (tid < MPAD) {
      float a = 0.0f;
      if (tid < MCAND) {
        if (it == 0) {
          a = u01(pcg_hash(s * 197u + (unsigned)tid * 13u + 0x9E3779B9u));
        } else {
          unsigned h1 = pcg_hash((s * CEM_ITERS + (unsigned)it) * 64u + (unsigned)tid);
          unsigned h2 = pcg_hash(h1 ^ 0x85EBCA6Bu);
          float u1v = fmaxf(u01(h1), 1.0e-12f);
          float nrm = sqrtf(-2.0f * __logf(u1v)) * __cosf(6.28318530718f * u01(h2));
          a = sMu + sStd * nrm;
        }
      }
      sAng[tid] = a;
    }
    if (tid == 0) { sSum = 0.0f; sSumSq = 0.0f; }
    __syncthreads();

    // ---- layer 1 (transposed): W1^T(16x4pad32) @ X^T(4pad32x16) per wave ----
    // B f16 32x16: lanes 0-15 hold col N=lane, K {0..7,16..23}; only K=0..3 != 0
    v16h xb = {};
    if (half == 0) {
      xb[0] = (_Float16)sx;
      xb[1] = (_Float16)sy;
      xb[2] = (_Float16)sAng[hrow];
      xb[3] = (_Float16)1.0f;            // bias lane
    }
    #pragma unroll
    for (int t = 0; t < NT; ++t) {
      const v16h af = *(const v16h*)&sW1Tp[t][lane][0];  // 2x ds_load_b128
      v8f c = {};
      c = __builtin_amdgcn_wmma_f32_16x16x32_f16(false, af, false, xb,
                                                 (short)0, c, false, false);
      // C: lane -> cand = hrow; VGPR r -> hidden n = 16t + 8*half + r.
      // 8 consecutive hidden units => one contiguous v8h store (kperm keeps
      // 8-aligned runs contiguous), XOR-swizzled by row for bank spread.
      v8h hv;
      #pragma unroll
      for (int r = 0; r < 8; ++r) {
        float v = c[r];
        v = v > 0.0f ? v : 0.0f;         // relu (bias already in WMMA)
        hv[r] = (_Float16)v;
      }
      const int pos = kperm(t * 16 + half * 8);
      const int col = ((pos >> 4) ^ rxor) * 16 + (pos & 15);
      *(v8h*)&sHp[hrow][col] = hv;       // 1x ds_store_b128
    }
    __syncthreads();

    // ---- layer 2: H(16x128) @ W2(128x112), K-outer so A frag is reused ----
    v8f acc[NT];
    #pragma unroll
    for (int t = 0; t < NT; ++t) acc[t] = (v8f){};
    #pragma unroll
    for (int ks = 0; ks < KT; ++ks) {
      const int g = ((ks * 2 + half) ^ rxor) * 16;       // swizzled group
      const v16h av = *(const v16h*)&sHp[hrow][g];       // 2x ds_load_b128
      #pragma unroll
      for (int t = 0; t < NT; ++t) {
        const v16h bf = *(const v16h*)&sW2p[t][ks][lane][0];
        acc[t] = __builtin_amdgcn_wmma_f32_16x16x32_f16(false, av, false, bf,
                                                        (short)0, acc[t],
                                                        false, false);
      }
    }
    // fused layer 3: q[m] = sum_n relu(h2[m,n] + b2[n]) * w3[n]  (+ b3)
    float qp[8];
    #pragma unroll
    for (int r = 0; r < 8; ++r) qp[r] = 0.0f;
    #pragma unroll
    for (int t = 0; t < NT; ++t) {
      const int   n   = t * 16 + nloc;
      const float w3n = sW3[n];
      const float bb  = sB2[n];
      #pragma unroll
      for (int r = 0; r < 8; ++r) {
        float v = acc[t][r] + bb;
        v = v > 0.0f ? v : 0.0f;
        qp[r] += v * w3n;
      }
    }
    // reduce partial q across the 16 lanes of each half (columns)
    #pragma unroll
    for (int r = 0; r < 8; ++r) {
      float v = qp[r];
      v += __shfl_xor(v, 1, 32);
      v += __shfl_xor(v, 2, 32);
      v += __shfl_xor(v, 4, 32);
      v += __shfl_xor(v, 8, 32);
      if (nloc == 0) sQ[base + half * 8 + r] = v + B3;
    }
    __syncthreads();

    // ---- top-32-of-50 via rank counting; mean + unbiased std of elites ----
    if (tid < MCAND) {
      float qi = sQ[tid];
      int rank = 0;
      for (int j = 0; j < MCAND; ++j) {
        float qj = sQ[j];
        rank += (qj > qi) || ((qj == qi) && (j < tid));
      }
      if (rank < NTOP) {
        atomicAdd(&sSum, qi);
        atomicAdd(&sSumSq, qi * qi);
      }
    }
    __syncthreads();
    if (tid == 0) {
      float mu  = sSum * (1.0f / NTOP);
      float var = (sSumSq - (float)NTOP * mu * mu) * (1.0f / (NTOP - 1));
      sMu  = mu;
      sStd = sqrtf(fmaxf(var, 0.0f));
    }
    __syncthreads();
  }

  if (tid == 0) out[s] = sMu * 6.28318530718f;   // mus[-1] * 2*pi
}

extern "C" void kernel_launch(void* const* d_in, const int* in_sizes, int n_in,
                              void* d_out, int out_size, void* d_ws, size_t ws_size,
                              hipStream_t stream) {
  const float* states = (const float*)d_in[0];
  const float* W1     = (const float*)d_in[1];
  const float* b1     = (const float*)d_in[2];
  const float* W2     = (const float*)d_in[3];
  const float* b2     = (const float*)d_in[4];
  const float* W3     = (const float*)d_in[5];
  const float* b3     = (const float*)d_in[6];
  float* out = (float*)d_out;
  const int batch = in_sizes[0] / 2;   // states is (batch, 2)
  cem_dqn_kernel<<<batch, 128, 0, stream>>>(states, W1, b1, W2, b2, W3, b3, out);
  (void)d_ws; (void)ws_size; (void)n_in; (void)out_size;
}
